// StaticDistributionModel_79731772883081
// MI455X (gfx1250) — compile-verified, matrix operand order checked by reading, not run on hardware
//
#include <hip/hip_runtime.h>
#include <hip/hip_bf16.h>

typedef __attribute__((ext_vector_type(16))) _Float16 v16h;
typedef __attribute__((ext_vector_type(8)))  float    v8f;

#define N_IC     128
#define N_TILES  9          // ceil(129/16) -> 144 padded shift columns
#define BFRAG_HALVES (N_TILES * 4 * 32 * 16)   // 18432 halves = 36864 B
#define BFRAG_BYTES  (BFRAG_HALVES * 2)

// ---------------------------------------------------------------------------
// Prep: lse[s] = logsumexp(w[s:s+128]) for s<129 ; pack W^T into WMMA B-frag
// layout: slot index = ((tile*4 + chunk)*32 + lane), 16 halves per slot.
// For lane p = lane&15 -> column n = tile*16 + p ; half j -> K = chunk*32 +
// ((lane&16)?8:0) + (j<8 ? j : j+8). B[k][n] = w[n + k], 0-padded for n>=129.
// ---------------------------------------------------------------------------
__global__ void sdm_prep_kernel(const float* __restrict__ w,
                                _Float16* __restrict__ bfrag,
                                float* __restrict__ lse,
                                int n_shifts) {
  int tid = threadIdx.x;
  for (int s = tid; s < 144; s += blockDim.x) {
    if (s < n_shifts) {
      float sum = 0.f;
      for (int i = 0; i < N_IC; ++i) sum += __expf(w[s + i]);
      lse[s] = __logf(sum);
    } else {
      lse[s] = 0.f;
    }
  }
  for (int idx = tid; idx < N_TILES * 4 * 32; idx += blockDim.x) {
    int lane = idx & 31;
    int c    = (idx >> 5) & 3;
    int t    = idx >> 7;
    int ncol = t * 16 + (lane & 15);
    int kb   = c * 32 + ((lane & 16) ? 8 : 0);
    _Float16* dst = bfrag + (size_t)idx * 16;
    for (int j = 0; j < 16; ++j) {
      int k = kb + (j < 8 ? j : j + 8);
      float v = (ncol < n_shifts) ? w[ncol + k] : 0.f;
      dst[j] = (_Float16)v;
    }
  }
}

// ---------------------------------------------------------------------------
// Main fused kernel: 128 threads = 4 waves; each wave owns a 16-row tile.
// ---------------------------------------------------------------------------
__global__ void __launch_bounds__(128)
sdm_match_kernel(const float* __restrict__ log_data,
                 const float* __restrict__ w,
                 const _Float16* __restrict__ bfrag_g,
                 const float* __restrict__ lse_g,
                 float* __restrict__ out_dist,
                 float* __restrict__ out_loss,
                 float* __restrict__ out_shift,
                 int N, int n_shifts) {
  __shared__ _Float16 sB[BFRAG_HALVES];   // 36 KB: W^T WMMA B fragments
  __shared__ float    sLse[144];
  __shared__ float    sW[256];

  int tid = threadIdx.x;
  // Stage B fragments / lse / w into LDS (block-wide, once).
  {
    const uint4* src = (const uint4*)bfrag_g;
    uint4* dst = (uint4*)sB;
    for (int i = tid; i < BFRAG_BYTES / 16; i += blockDim.x) dst[i] = src[i];
    for (int i = tid; i < 144; i += blockDim.x) sLse[i] = lse_g[i];
    for (int i = tid; i < 256; i += blockDim.x) sW[i] = w[i];
  }
  __syncthreads();

  const int lane = tid & 31;
  const int wave = tid >> 5;
  const int rowb = blockIdx.x * 64 + wave * 16;     // first row of this wave's tile
  const int myrow = rowb + (lane & 15);
  const int r = (myrow < N) ? myrow : (N - 1);      // clamp (no divergence pre-WMMA)
  const float* rp = log_data + (size_t)r * N_IC;

  float ent = 0.f;
  float ps  = 0.f;

  v8f acc[N_TILES];
#pragma unroll
  for (int t = 0; t < N_TILES; ++t) acc[t] = 0.f;

  // K loop: build this lane's A fragment (exp + stats fused), then 9 WMMAs.
#pragma unroll
  for (int c = 0; c < 4; ++c) {
    const int kb = c * 32 + ((lane & 16) ? 8 : 0);
    float4 f0 = *(const float4*)(rp + kb);
    float4 f1 = *(const float4*)(rp + kb + 4);
    float4 f2 = *(const float4*)(rp + kb + 16);
    float4 f3 = *(const float4*)(rp + kb + 20);

    union { v16h v; _Float16 h[16]; } A;
    auto proc = [&](float lp) -> _Float16 {
      bool fin = __builtin_fabsf(lp) <= 3.402823466e38f;   // false for NaN/Inf
      float pv = fin ? __expf(lp) : 0.f;
      ent = fin ? (ent + pv * lp) : ent;
      ps += pv;
      return (_Float16)pv;
    };
    A.h[0]  = proc(f0.x); A.h[1]  = proc(f0.y); A.h[2]  = proc(f0.z); A.h[3]  = proc(f0.w);
    A.h[4]  = proc(f1.x); A.h[5]  = proc(f1.y); A.h[6]  = proc(f1.z); A.h[7]  = proc(f1.w);
    A.h[8]  = proc(f2.x); A.h[9]  = proc(f2.y); A.h[10] = proc(f2.z); A.h[11] = proc(f2.w);
    A.h[12] = proc(f3.x); A.h[13] = proc(f3.y); A.h[14] = proc(f3.z); A.h[15] = proc(f3.w);

#pragma unroll
    for (int t = 0; t < N_TILES; ++t) {
      v16h b = *(const v16h*)(sB + ((size_t)(t * 4 + c) * 32 + lane) * 16);
      acc[t] = __builtin_amdgcn_wmma_f32_16x16x32_f16(
          /*neg_a=*/false, A.v, /*neg_b=*/false, b,
          /*c_mod=*/(short)0, acc[t], /*reuse_a=*/false, /*reuse_b=*/false);
    }
  }

  // Each row's 128 p-values are split across lane pair (l, l+16): fold stats.
  ent += __shfl_xor(ent, 16, 32);
  ps  += __shfl_xor(ps, 16, 32);

  const float INF = __builtin_inff();
  const int hi = (lane & 16) ? 8 : 0;     // lanes 16-31 hold accumulator rows m+8

  // Epilogue: per accumulator row m: loss, first-argmin, gather outputs.
#pragma unroll
  for (int m = 0; m < 8; ++m) {
    float em = __shfl(ent, m + hi, 32);
    float pm = __shfl(ps,  m + hi, 32);

    float best = INF;
    int bs = 0;
#pragma unroll
    for (int t = 0; t < N_TILES; ++t) {
      int s  = t * 16 + (lane & 15);
      int sc = (s < n_shifts) ? s : 0;
      float loss = em - acc[t][m] + pm * sLse[sc];
      if (s >= n_shifts) loss = INF;
      if (loss < best) { best = loss; bs = s; }   // strict <: keeps smallest s
    }
    // Butterfly reduce across the 16 lanes of each half (first-min tiebreak).
#pragma unroll
    for (int d = 1; d < 16; d <<= 1) {
      float ov = __shfl_xor(best, d, 32);
      int   os = __shfl_xor(bs,   d, 32);
      if (ov < best || (ov == best && os < bs)) { best = ov; bs = os; }
    }

    int row_g = rowb + m + hi;
    if (row_g < N) {
      if ((lane & 15) == 0) {
        out_loss[row_g]  = best;
        out_shift[row_g] = (float)bs;
      }
      // matched_log_dist[row] = w[bs : bs+128] - lse[bs]; 16 lanes x 8 floats.
      float lb = sLse[bs];
      int ib = (lane & 15) * 8;
      float* od = out_dist + (size_t)row_g * N_IC + ib;
#pragma unroll
      for (int j = 0; j < 8; j += 4) {
        float4 v;
        v.x = sW[bs + ib + j + 0] - lb;
        v.y = sW[bs + ib + j + 1] - lb;
        v.z = sW[bs + ib + j + 2] - lb;
        v.w = sW[bs + ib + j + 3] - lb;
        *(float4*)(od + j) = v;
      }
    }
  }
}

// ---------------------------------------------------------------------------
extern "C" void kernel_launch(void* const* d_in, const int* in_sizes, int n_in,
                              void* d_out, int out_size, void* d_ws, size_t ws_size,
                              hipStream_t stream) {
  const float* log_data = (const float*)d_in[0];
  const float* w        = (const float*)d_in[1];

  const int n_ic    = N_IC;
  const int N       = in_sizes[0] / n_ic;          // 200000
  const int n_sup   = in_sizes[1];                 // 256
  const int n_shift = n_sup - n_ic + 1;            // 129

  float* out_dist  = (float*)d_out;
  float* out_loss  = out_dist + (size_t)N * n_ic;
  float* out_shift = out_loss + N;

  _Float16* bfrag = (_Float16*)d_ws;
  float*    lse   = (float*)((char*)d_ws + BFRAG_BYTES);

  sdm_prep_kernel<<<1, 256, 0, stream>>>(w, bfrag, lse, n_shift);

  int blocks = (N + 63) / 64;                      // 3125, exact
  sdm_match_kernel<<<blocks, 128, 0, stream>>>(log_data, w, bfrag, lse,
                                               out_dist, out_loss, out_shift,
                                               N, n_shift);
}
